// EncoderLayer_35407710389142
// MI455X (gfx1250) — matrix operand order repaired and checked by reading, not compiled, not run on hardware
//
#include <hip/hip_runtime.h>

// ---------------------------------------------------------------------------
// CDNA5 (gfx1250) wave32 WMMA encoder layer, async-to-LDS pipelined.
// B=4, T=1024, D=1024, H=16, HD=64, DFF=4096, NAX=1, DPR=32
// ---------------------------------------------------------------------------

typedef __attribute__((ext_vector_type(16))) __bf16 v16bf;
typedef __attribute__((ext_vector_type(8)))  float  v8f;
typedef __attribute__((ext_vector_type(4)))  unsigned int u32x4;

static __device__ __forceinline__ v8f wmma_bf16(v16bf a, v16bf b, v8f c) {
  return __builtin_amdgcn_wmma_f32_16x16x32_bf16(false, a, false, b, (short)0, c,
                                                 false, false);
}

// Async global->LDS 128-bit copy (CDNA5 GLOBAL_LOAD_ASYNC_TO_LDS_B128).
// ldsaddr = wave-relative LDS byte offset (low 32 bits of generic shared ptr).
static __device__ __forceinline__ void async_b128(unsigned ldsaddr, const void* gaddr) {
  asm volatile("global_load_async_to_lds_b128 %0, %1, off"
               :: "v"(ldsaddr), "v"(gaddr) : "memory");
}
static __device__ __forceinline__ void wait_async0() {
  asm volatile("s_wait_asynccnt 0x0" ::: "memory");
}
static __device__ __forceinline__ unsigned lds_off(const void* p) {
  return (unsigned)(size_t)p;   // LDS aperture: addr[31:0] is the LDS offset
}

// A fragment (16x32 bf16, MxK). Lane l: M = l&15, K-base = (l<16)?0:8.
// Two contiguous 8-element runs at p and p+16 -> 2x ds_load_b128.
static __device__ __forceinline__ v16bf load_frag_a(const __bf16* src, int ld) {
  int lane = threadIdx.x & 31;
  const __bf16* p = src + (size_t)(lane & 15) * ld + ((lane >> 4) << 3);
  union { u32x4 u[2]; v16bf v; } t;
  t.u[0] = *(const u32x4*)(p);
  t.u[1] = *(const u32x4*)(p + 16);
  return t.v;
}

// B fragment (32x16 bf16, KxN), n-major storage src[n*ld + k].
// Lane l: N = l&15, K-base = (l<16)?0:8 ; two 8-element runs at p and p+16.
static __device__ __forceinline__ v16bf load_frag_b(const __bf16* src, int ld) {
  int lane = threadIdx.x & 31;
  const __bf16* p = src + (size_t)(lane & 15) * ld + ((lane >> 4) << 3);
  union { u32x4 u[2]; v16bf v; } t;
  t.u[0] = *(const u32x4*)(p);
  t.u[1] = *(const u32x4*)(p + 16);
  return t.v;
}

// ---------------------------------------------------------------------------
// Weight transpose + fp32->bf16: src[K][N] fp32 -> dst[N][K] bf16 (LDS tiled).
// ---------------------------------------------------------------------------
__global__ __launch_bounds__(256) void transpose_bf16_kernel(const float* __restrict__ src,
                                                             __bf16* __restrict__ dst,
                                                             int K, int N) {
  __shared__ __bf16 tile[32][33];
  int bx = blockIdx.x * 32;                    // n
  int by = blockIdx.y * 32;                    // k
  int tx = threadIdx.x & 31, ty = threadIdx.x >> 5;   // ty 0..7
  #pragma unroll
  for (int i = 0; i < 32; i += 8)
    tile[ty + i][tx] = (__bf16)src[(size_t)(by + ty + i) * N + bx + tx];
  __syncthreads();
  #pragma unroll
  for (int i = 0; i < 32; i += 8)
    dst[(size_t)(bx + ty + i) * K + by + tx] = tile[tx][ty + i];
}

// ---------------------------------------------------------------------------
// LayerNorm (optional fused residual x = x1 + x2); bf16 h out, optional fp32 x.
// ---------------------------------------------------------------------------
__global__ __launch_bounds__(256) void ln_kernel(const float* __restrict__ x1,
                                                 const float* __restrict__ x2,
                                                 const float* __restrict__ w,
                                                 const float* __restrict__ b,
                                                 __bf16* __restrict__ hout,
                                                 float* __restrict__ xout) {
  const int D = 1024;
  __shared__ float red[256];
  int row = blockIdx.x, t = threadIdx.x;
  const float* p1 = x1 + (size_t)row * D;
  const float* p2 = x2 ? x2 + (size_t)row * D : nullptr;
  float v[4];
  float s = 0.f;
  #pragma unroll
  for (int i = 0; i < 4; ++i) {
    float a = p1[t + 256 * i];
    if (p2) a += p2[t + 256 * i];
    v[i] = a; s += a;
  }
  red[t] = s; __syncthreads();
  for (int st = 128; st > 0; st >>= 1) { if (t < st) red[t] += red[t + st]; __syncthreads(); }
  float mu = red[0] * (1.f / D);
  __syncthreads();
  float sq = 0.f;
  #pragma unroll
  for (int i = 0; i < 4; ++i) { float d = v[i] - mu; sq += d * d; }
  red[t] = sq; __syncthreads();
  for (int st = 128; st > 0; st >>= 1) { if (t < st) red[t] += red[t + st]; __syncthreads(); }
  float rstd = rsqrtf(red[0] * (1.f / D) + 1e-5f);
  #pragma unroll
  for (int i = 0; i < 4; ++i) {
    int col = t + 256 * i;
    float hn = (v[i] - mu) * rstd * w[col] + b[col];
    hout[(size_t)row * D + col] = (__bf16)hn;
    if (xout) xout[(size_t)row * D + col] = v[i];
  }
}

// ---------------------------------------------------------------------------
// TN GEMM: C[M,N] = A[M,K] @ Bt[N,K]^T (+bias,+resid), fp32 out.
// Double-buffered LDS fed by async global->LDS b128; 8 waves x 64x32 tiles.
// ---------------------------------------------------------------------------
#define BM 128
#define BN 128
#define BK 32

static __device__ __forceinline__ void issue_tile2(const __bf16* Ag, const __bf16* Bg,
                                                   __bf16* sa, __bf16* sb, int K, int tid) {
  int r0 = tid >> 2,          o0 = (tid & 3) << 3;          // chunk tid
  int r1 = (tid + 256) >> 2,  o1 = (tid & 3) << 3;          // chunk tid+256
  async_b128(lds_off(sa + r0 * BK + o0), Ag + (size_t)r0 * K + o0);
  async_b128(lds_off(sa + r1 * BK + o1), Ag + (size_t)r1 * K + o1);
  async_b128(lds_off(sb + r0 * BK + o0), Bg + (size_t)r0 * K + o0);
  async_b128(lds_off(sb + r1 * BK + o1), Bg + (size_t)r1 * K + o1);
}

__global__ __launch_bounds__(256) void gemm_tn_kernel(
    const __bf16* __restrict__ A, const __bf16* __restrict__ Bt,
    float* __restrict__ C, const float* __restrict__ bias,
    const float* __restrict__ resid, int M, int N, int K) {
  __shared__ alignas(16) __bf16 sA[2][BM * BK];
  __shared__ alignas(16) __bf16 sB[2][BN * BK];
  int bn = blockIdx.x * BN, bm = blockIdx.y * BM;
  int tid = threadIdx.x;
  int w = tid >> 5, wr = w >> 2, wc = w & 3;
  const __bf16* Ag = A + (size_t)bm * K;
  const __bf16* Bg = Bt + (size_t)bn * K;
  v8f acc[4][2] = {};

  issue_tile2(Ag, Bg, sA[0], sB[0], K, tid);
  wait_async0();
  __syncthreads();
  int buf = 0;
  for (int k0 = 0; k0 < K; k0 += BK) {
    if (k0 + BK < K)
      issue_tile2(Ag + k0 + BK, Bg + k0 + BK, sA[buf ^ 1], sB[buf ^ 1], K, tid);
    #pragma unroll
    for (int ti = 0; ti < 4; ++ti) {
      v16bf a = load_frag_a(&sA[buf][(wr * 64 + ti * 16) * BK], BK);
      #pragma unroll
      for (int tj = 0; tj < 2; ++tj) {
        v16bf bf = load_frag_b(&sB[buf][(wc * 32 + tj * 16) * BK], BK);
        acc[ti][tj] = wmma_bf16(a, bf, acc[ti][tj]);
      }
    }
    wait_async0();
    __syncthreads();
    buf ^= 1;
  }
  int lane = tid & 31, col = lane & 15, rh = (lane >> 4) << 3;
  #pragma unroll
  for (int ti = 0; ti < 4; ++ti)
    #pragma unroll
    for (int tj = 0; tj < 2; ++tj) {
      int gcol = bn + wc * 32 + tj * 16 + col;
      float bb = bias ? bias[gcol] : 0.f;
      #pragma unroll
      for (int v = 0; v < 8; ++v) {
        int grow = bm + wr * 64 + ti * 16 + rh + v;
        float r = acc[ti][tj][v] + bb;
        if (resid) r += resid[(size_t)grow * N + gcol];
        C[(size_t)grow * N + gcol] = r;
      }
    }
}

// ---------------------------------------------------------------------------
// fc1 TN GEMM + fused SwiGLU. Bt: [8192][1024] (transposed fc1_w).
// a-rows = bn+n ; gate-rows = 4096+bn+n. Writes silu(gate)*a as bf16.
// ---------------------------------------------------------------------------
__global__ __launch_bounds__(256) void gemm_swiglu_kernel(
    const __bf16* __restrict__ A, const __bf16* __restrict__ Bt,
    const float* __restrict__ bias, __bf16* __restrict__ G) {
  const int K = 1024, NG = 4096;
  __shared__ alignas(16) __bf16 sA[2][BM * BK];
  __shared__ alignas(16) __bf16 sBa[2][BN * BK];
  __shared__ alignas(16) __bf16 sBg[2][BN * BK];
  int bn = blockIdx.x * BN, bm = blockIdx.y * BM;
  int tid = threadIdx.x;
  int w = tid >> 5, wr = w >> 2, wc = w & 3;
  const __bf16* Ag  = A  + (size_t)bm * K;
  const __bf16* Bga = Bt + (size_t)bn * K;
  const __bf16* Bgg = Bt + (size_t)(NG + bn) * K;
  v8f accA[4][2] = {}, accG[4][2] = {};

  issue_tile2(Ag, Bga, sA[0], sBa[0], K, tid);
  {
    int r0 = tid >> 2, o0 = (tid & 3) << 3;
    int r1 = (tid + 256) >> 2;
    async_b128(lds_off(sBg[0] + r0 * BK + o0), Bgg + (size_t)r0 * K + o0);
    async_b128(lds_off(sBg[0] + r1 * BK + o0), Bgg + (size_t)r1 * K + o0);
  }
  wait_async0();
  __syncthreads();
  int buf = 0;
  for (int k0 = 0; k0 < K; k0 += BK) {
    if (k0 + BK < K) {
      issue_tile2(Ag + k0 + BK, Bga + k0 + BK, sA[buf ^ 1], sBa[buf ^ 1], K, tid);
      int r0 = tid >> 2, o0 = (tid & 3) << 3;
      int r1 = (tid + 256) >> 2;
      async_b128(lds_off(sBg[buf ^ 1] + r0 * BK + o0), Bgg + k0 + BK + (size_t)r0 * K + o0);
      async_b128(lds_off(sBg[buf ^ 1] + r1 * BK + o0), Bgg + k0 + BK + (size_t)r1 * K + o0);
    }
    #pragma unroll
    for (int ti = 0; ti < 4; ++ti) {
      v16bf a = load_frag_a(&sA[buf][(wr * 64 + ti * 16) * BK], BK);
      #pragma unroll
      for (int tj = 0; tj < 2; ++tj) {
        v16bf b1 = load_frag_b(&sBa[buf][(wc * 32 + tj * 16) * BK], BK);
        accA[ti][tj] = wmma_bf16(a, b1, accA[ti][tj]);
        v16bf b2 = load_frag_b(&sBg[buf][(wc * 32 + tj * 16) * BK], BK);
        accG[ti][tj] = wmma_bf16(a, b2, accG[ti][tj]);
      }
    }
    wait_async0();
    __syncthreads();
    buf ^= 1;
  }
  int lane = tid & 31, col = lane & 15, rh = (lane >> 4) << 3;
  #pragma unroll
  for (int ti = 0; ti < 4; ++ti)
    #pragma unroll
    for (int tj = 0; tj < 2; ++tj) {
      int gcol = bn + wc * 32 + tj * 16 + col;
      float ba = bias[gcol], bg = bias[gcol + NG];
      #pragma unroll
      for (int v = 0; v < 8; ++v) {
        int grow = bm + wr * 64 + ti * 16 + rh + v;
        float fa = accA[ti][tj][v] + ba;
        float fg = accG[ti][tj][v] + bg;
        float sw = fg / (1.f + __expf(-fg));
        G[(size_t)grow * NG + gcol] = (__bf16)(sw * fa);
      }
    }
}

// ---------------------------------------------------------------------------
// Type-embedding add + interleaved RoPE (first 32 of 64 dims per head),
// fp32 QKV -> bf16 Q/K/V. One thread per (token, dim-pair).
// ---------------------------------------------------------------------------
__global__ __launch_bounds__(256) void rope_kernel(
    const float* __restrict__ qkv, const float* __restrict__ type_emb,
    const int* __restrict__ x_type, const int* __restrict__ seq_order,
    __bf16* __restrict__ Qb, __bf16* __restrict__ Kb, __bf16* __restrict__ Vb) {
  int idx = blockIdx.x * 256 + threadIdx.x;
  int row = idx >> 9;
  int d = (idx & 511) * 2;
  const float* base = qkv + (size_t)row * 3072;
  const float* te = type_emb + (size_t)x_type[row] * 2048;
  float q0 = base[d] + te[d],               q1 = base[d + 1] + te[d + 1];
  float k0 = base[1024 + d] + te[1024 + d], k1 = base[1025 + d] + te[1025 + d];
  float v0 = base[2048 + d],                v1 = base[2049 + d];
  int dl = d & 63;
  if (dl < 32) {
    float pos = (float)seq_order[row];
    float inv = __powf(10000.f, -(float)dl * (1.f / 32.f));
    float th = pos * inv;
    float sn, cs;
    __sincosf(th, &sn, &cs);
    float nq0 = q0 * cs - q1 * sn, nq1 = q1 * cs + q0 * sn;
    float nk0 = k0 * cs - k1 * sn, nk1 = k1 * cs + k0 * sn;
    q0 = nq0; q1 = nq1; k0 = nk0; k1 = nk1;
  }
  size_t o = (size_t)row * 1024 + d;
  Qb[o] = (__bf16)q0; Qb[o + 1] = (__bf16)q1;
  Kb[o] = (__bf16)k0; Kb[o + 1] = (__bf16)k1;
  Vb[o] = (__bf16)v0; Vb[o + 1] = (__bf16)v1;
}

// ---------------------------------------------------------------------------
// Flash attention: block = (b,h,128 q-rows); wave = 16 q-rows.
// K tile streamed via async-to-LDS; S = Q@K^T chained WMMA; online softmax;
// P round-trips wave-private LDS; O += P@V via WMMA.
// ---------------------------------------------------------------------------
__global__ __launch_bounds__(256) void attn_kernel(const __bf16* __restrict__ Qb,
                                                   const __bf16* __restrict__ Kb,
                                                   const __bf16* __restrict__ Vb,
                                                   float* __restrict__ O) {
  __shared__ alignas(16) __bf16 sK[32 * 64];
  __shared__ alignas(16) __bf16 sVt[64 * 32];
  __shared__ alignas(16) __bf16 sP[8][16 * 32];
  int blk = blockIdx.x;
  int qblk = blk & 7, bh = blk >> 3;
  int b = bh >> 4, h = bh & 15;
  const size_t hb = ((size_t)b * 1024) * 1024 + h * 64;
  const __bf16* Qp = Qb + hb;
  const __bf16* Kp = Kb + hb;
  const __bf16* Vp = Vb + hb;
  int w = threadIdx.x >> 5, lane = threadIdx.x & 31;
  int qr0 = qblk * 128 + w * 16;
  v16bf aq0 = load_frag_a(Qp + (size_t)qr0 * 1024, 1024);
  v16bf aq1 = load_frag_a(Qp + (size_t)qr0 * 1024 + 32, 1024);
  v8f o0 = {}, o1 = {}, o2 = {}, o3 = {};
  float mrow[8], lrow[8];
  #pragma unroll
  for (int i = 0; i < 8; ++i) { mrow[i] = -1e30f; lrow[i] = 0.f; }

  int key = threadIdx.x >> 3, doff = (threadIdx.x & 7) * 8;
  for (int kb0 = 0; kb0 < 1024; kb0 += 32) {
    // K tile: async global->LDS (contiguous b128 per thread)
    async_b128(lds_off(&sK[key * 64 + doff]), Kp + (size_t)(kb0 + key) * 1024 + doff);
    // V tile: transpose into [d][key] via VGPR
    {
      union { u32x4 u; __bf16 hh[8]; } tv;
      tv.u = *(const u32x4*)(Vp + (size_t)(kb0 + key) * 1024 + doff);
      #pragma unroll
      for (int i = 0; i < 8; ++i) sVt[(doff + i) * 32 + key] = tv.hh[i];
    }
    wait_async0();
    __syncthreads();

    v8f s0 = {}, s1 = {};
    v16bf bk;
    bk = load_frag_b(sK + 32, 64);           s0 = wmma_bf16(aq1, bk, s0);
    bk = load_frag_b(sK, 64);                s0 = wmma_bf16(aq0, bk, s0);
    bk = load_frag_b(sK + 16 * 64 + 32, 64); s1 = wmma_bf16(aq1, bk, s1);
    bk = load_frag_b(sK + 16 * 64, 64);      s1 = wmma_bf16(aq0, bk, s1);

    #pragma unroll
    for (int v = 0; v < 8; ++v) {
      float a0 = s0[v] * 0.125f, a1 = s1[v] * 0.125f;
      float mx = fmaxf(a0, a1);
      #pragma unroll
      for (int off = 1; off < 16; off <<= 1) mx = fmaxf(mx, __shfl_xor(mx, off, 32));
      float mn = fmaxf(mrow[v], mx);
      float corr = __expf(mrow[v] - mn);
      float p0 = __expf(a0 - mn), p1 = __expf(a1 - mn);
      float rs = p0 + p1;
      #pragma unroll
      for (int off = 1; off < 16; off <<= 1) rs += __shfl_xor(rs, off, 32);
      lrow[v] = lrow[v] * corr + rs;
      mrow[v] = mn;
      o0[v] *= corr; o1[v] *= corr; o2[v] *= corr; o3[v] *= corr;
      int prow = v + ((lane >> 4) << 3);
      sP[w][prow * 32 + (lane & 15)] = (__bf16)p0;
      sP[w][prow * 32 + 16 + (lane & 15)] = (__bf16)p1;
    }
    asm volatile("s_wait_dscnt 0x0" ::: "memory");   // wave-local P store->load

    v16bf ap = load_frag_a(&sP[w][0], 32);
    v16bf bv;
    bv = load_frag_b(sVt + 0 * 16 * 32, 32); o0 = wmma_bf16(ap, bv, o0);
    bv = load_frag_b(sVt + 1 * 16 * 32, 32); o1 = wmma_bf16(ap, bv, o1);
    bv = load_frag_b(sVt + 2 * 16 * 32, 32); o2 = wmma_bf16(ap, bv, o2);
    bv = load_frag_b(sVt + 3 * 16 * 32, 32); o3 = wmma_bf16(ap, bv, o3);
    __syncthreads();
  }

  int col = lane & 15, rh = (lane >> 4) << 3;
  float* Op = O + ((size_t)b * 1024 + qr0) * 1024 + h * 64;
  #pragma unroll
  for (int v = 0; v < 8; ++v) {
    float inv = 1.f / lrow[v];
    size_t r = (size_t)(rh + v) * 1024;
    Op[r + col]      = o0[v] * inv;
    Op[r + 16 + col] = o1[v] * inv;
    Op[r + 32 + col] = o2[v] * inv;
    Op[r + 48 + col] = o3[v] * inv;
  }
}

// ---------------------------------------------------------------------------
// Host-side orchestration
// ---------------------------------------------------------------------------
extern "C" void kernel_launch(void* const* d_in, const int* in_sizes, int n_in,
                              void* d_out, int out_size, void* d_ws, size_t ws_size,
                              hipStream_t stream) {
  (void)in_sizes; (void)n_in; (void)out_size; (void)ws_size;
  const int*   x_type    = (const int*)  d_in[0];
  const float* x_value   = (const float*)d_in[1];
  const int*   seq_order = (const int*)  d_in[2];
  const float* Wqkv      = (const float*)d_in[3];
  const float* type_emb  = (const float*)d_in[4];
  const float* ln1_w     = (const float*)d_in[5];
  const float* ln1_b     = (const float*)d_in[6];
  const float* ln2_w     = (const float*)d_in[7];
  const float* ln2_b     = (const float*)d_in[8];
  const float* fc1_w     = (const float*)d_in[9];
  const float* fc1_b     = (const float*)d_in[10];
  const float* fc2_w     = (const float*)d_in[11];
  const float* fc2_b     = (const float*)d_in[12];
  float* out = (float*)d_out;

  char* wsp = (char*)d_ws;
  size_t off = 0;
  auto take = [&](size_t bytes) -> void* {
    void* p = wsp + off;
    off += (bytes + 255) & ~(size_t)255;
    return p;
  };
  __bf16* wqkvT = (__bf16*)take((size_t)3072 * 1024 * 2);  // [3072][1024]
  __bf16* fc1T  = (__bf16*)take((size_t)8192 * 1024 * 2);  // [8192][1024]
  __bf16* fc2T  = (__bf16*)take((size_t)1024 * 4096 * 2);  // [1024][4096]
  __bf16* h1    = (__bf16*)take((size_t)4096 * 1024 * 2);
  float*  qkvF  = (float*) take((size_t)4096 * 3072 * 4);
  __bf16* qB    = (__bf16*)take((size_t)4096 * 1024 * 2);
  __bf16* kB    = (__bf16*)take((size_t)4096 * 1024 * 2);
  __bf16* vB    = (__bf16*)take((size_t)4096 * 1024 * 2);
  float*  Obuf  = (float*) take((size_t)4096 * 1024 * 4);
  float*  xbuf  = (float*) take((size_t)4096 * 1024 * 4);
  __bf16* h2    = (__bf16*)take((size_t)4096 * 1024 * 2);
  __bf16* gB    = (__bf16*)qkvF;   // reuse dead QKV-fp32 region (32MB <= 48MB)

  // 1) weights -> bf16, transposed to [N][K]
  transpose_bf16_kernel<<<dim3(3072 / 32, 1024 / 32), 256, 0, stream>>>(Wqkv, wqkvT, 1024, 3072);
  transpose_bf16_kernel<<<dim3(8192 / 32, 1024 / 32), 256, 0, stream>>>(fc1_w, fc1T, 1024, 8192);
  transpose_bf16_kernel<<<dim3(1024 / 32, 4096 / 32), 256, 0, stream>>>(fc2_w, fc2T, 4096, 1024);
  // 2) LN1
  ln_kernel<<<4096, 256, 0, stream>>>(x_value, nullptr, ln1_w, ln1_b, h1, nullptr);
  // 3) QKV GEMM  (4096 x 3072 x 1024)
  gemm_tn_kernel<<<dim3(3072 / BN, 4096 / BM), 256, 0, stream>>>(
      h1, wqkvT, qkvF, nullptr, nullptr, 4096, 3072, 1024);
  // 4) type-emb + RoPE -> bf16 Q/K/V
  rope_kernel<<<(4096 * 512) / 256, 256, 0, stream>>>(qkvF, type_emb, x_type,
                                                      seq_order, qB, kB, vB);
  // 5) flash attention
  attn_kernel<<<512, 256, 0, stream>>>(qB, kB, vB, Obuf);
  // 6) x = x_value + O ; LN2
  ln_kernel<<<4096, 256, 0, stream>>>(x_value, Obuf, ln2_w, ln2_b, h2, xbuf);
  // 7) fc1 + SwiGLU  (4096 x 4096(,+4096) x 1024)
  gemm_swiglu_kernel<<<dim3(4096 / BN, 4096 / BM), 256, 0, stream>>>(h2, fc1T, fc1_b, gB);
  // 8) fc2 + bias + residual -> out  (4096 x 1024 x 4096)
  gemm_tn_kernel<<<dim3(1024 / BN, 4096 / BM), 256, 0, stream>>>(
      gB, fc2T, out, fc2_b, xbuf, 4096, 1024, 4096);
}
